// FloatingPointHGRNLayer_78950088835411
// MI455X (gfx1250) — compile-verified, matrix operand order checked by reading, not run on hardware
//
#include <hip/hip_runtime.h>
#include <hip/hip_bf16.h>

// ---------------- problem constants ----------------
#define Bn   4
#define Sn   2048
#define Hn   1024
#define BSn  (Bn * Sn)          // 8192 flattened rows
#define CHUNK 128
#define NC   (Sn / CHUNK)       // 16 chunks per sequence
#define NT_N (Hn / 16)          // 64 n-tiles
#define NT_M (BSn / 16)         // 512 m-tiles

typedef __bf16 v16bf  __attribute__((ext_vector_type(16)));
typedef __bf16 bf16x8 __attribute__((ext_vector_type(8)));
typedef float  v8f    __attribute__((ext_vector_type(8)));

// ---------------- workspace layout (bytes) ----------------
#define OFF_F   ((size_t)0)            // f = sigmoid(x@Wf^T)  fp32 [BS,H]  32MB
#define OFF_I   ((size_t)33554432)     // inp                  fp32 [BS,H]  32MB
#define OFF_G   ((size_t)67108864)     // g                    bf16 [BS,H]  16MB
#define OFF_U   ((size_t)83886080)     // u = g*h              bf16 [BS,H]  16MB
#define OFF_XB  ((size_t)100663296)    // x in bf16            [BS,H]       16MB
#define OFF_WFB ((size_t)117440512)    // Wf bf16 [H,H]  2MB
#define OFF_WIB ((size_t)119537664)    // Wi bf16
#define OFF_WGB ((size_t)121634816)    // Wg bf16
#define OFF_WOB ((size_t)123731968)    // Wo bf16
#define OFF_FP  ((size_t)125829120)    // chunk prod(f)  fp32 [B,NC,H] 256KB
#define OFF_HE  ((size_t)126091264)    // chunk h_end
#define OFF_CA  ((size_t)126353408)    // chunk carry-in

// ---------------- fp32 -> bf16 convert pre-pass ----------------
__global__ __launch_bounds__(256) void cvt_bf16(const float* __restrict__ src,
                                                __bf16* __restrict__ dst) {
  const size_t i = ((size_t)blockIdx.x * 256 + threadIdx.x) * 8;
  float4 a = *(const float4*)(src + i);
  float4 b = *(const float4*)(src + i + 4);
  bf16x8 o;
  o[0] = (__bf16)a.x; o[1] = (__bf16)a.y; o[2] = (__bf16)a.z; o[3] = (__bf16)a.w;
  o[4] = (__bf16)b.x; o[5] = (__bf16)b.y; o[6] = (__bf16)b.z; o[7] = (__bf16)b.w;
  *(bf16x8*)(dst + i) = o;
}

// ---------------- WMMA fragment loader (bf16, b128 loads) ----------------
// A-matrix 16x32 bf16 layout (ISA 7.12.2): lanes 0-15 hold row M=lane with
// K 0..7 (elems 0..7) and K 16..23 (elems 8..15); lanes 16-31 hold the same
// row with K 8..15 / 24..31.  B = W^T -> lane n reads row n of W contiguously,
// so A and B fragment addressing is identical.
__device__ __forceinline__ v16bf load_frag(const __bf16* __restrict__ base,
                                           int row0, int k0, int lane) {
  const int r    = lane & 15;
  const int half = lane >> 4;
  const __bf16* p = base + (size_t)(row0 + r) * Hn + k0 + half * 8;
  bf16x8 lo = *(const bf16x8*)(p);        // 16B aligned -> global_load_b128
  bf16x8 hi = *(const bf16x8*)(p + 16);
  return __builtin_shufflevector(lo, hi, 0, 1, 2, 3, 4, 5, 6, 7,
                                         8, 9, 10, 11, 12, 13, 14, 15);
}

__device__ __forceinline__ v8f wmma_bf16(v16bf a, v16bf b, v8f c) {
  return __builtin_amdgcn_wmma_f32_16x16x32_bf16(false, a, false, b,
                                                 (short)0, c, false, false);
}

__device__ __forceinline__ float fast_sigmoid(float x) {
  return 1.0f / (1.0f + __expf(-x));
}

// fused epilogue for one 16x16 tile of the triple projection
__device__ __forceinline__ void store_fig(const v8f& pF, const v8f& pI,
                                          const v8f& pG, int mrow, int ncol,
                                          int lane, float* __restrict__ F,
                                          float* __restrict__ INP,
                                          __bf16* __restrict__ G) {
  const int r    = lane & 15;
  const int half = lane >> 4;
#pragma unroll
  for (int j = 0; j < 8; ++j) {
    const size_t idx = (size_t)(mrow + half * 8 + j) * Hn + ncol + r;
    float f  = fast_sigmoid(pF[j]);
    float pi = pI[j];
    F[idx]   = f;
    INP[idx] = pi * fast_sigmoid(pi) * (1.0f - f);
    G[idx]   = (__bf16)pG[j];
  }
}

// ---------------- kernel 1: fused triple projection ----------------
// Wave tile: 32 rows x 32 cols x 3 weights (12 accumulators).  Each k-step:
// 2 A frags + 6 B frags (16 b128 loads) feed 12 v_wmma -> 1.33 loads/wmma.
__global__ __launch_bounds__(128) void hgrn_proj3(
    const __bf16* __restrict__ Xb, const __bf16* __restrict__ Wfb,
    const __bf16* __restrict__ Wib, const __bf16* __restrict__ Wgb,
    float* __restrict__ F, float* __restrict__ INP, __bf16* __restrict__ G) {
  const int lane = threadIdx.x & 31;
  const int wave = threadIdx.x >> 5;
  const int tile = blockIdx.x * 4 + wave;        // over (m-pair, n-pair)
  const int NPAIR = NT_N / 2;                    // 32
  const int mt = tile / NPAIR;
  const int np = tile - mt * NPAIR;
  const int mrow0 = mt * 32;
  const int mrow1 = mrow0 + 16;
  const int ncol0 = np * 32;
  const int ncol1 = ncol0 + 16;

  v8f f00 = {}, f01 = {}, f10 = {}, f11 = {};
  v8f i00 = {}, i01 = {}, i10 = {}, i11 = {};
  v8f g00 = {}, g01 = {}, g10 = {}, g11 = {};
  for (int k0 = 0; k0 < Hn; k0 += 32) {
    v16bf a0 = load_frag(Xb, mrow0, k0, lane);
    v16bf a1 = load_frag(Xb, mrow1, k0, lane);
    v16bf b;
    b = load_frag(Wfb, ncol0, k0, lane);
    f00 = wmma_bf16(a0, b, f00); f10 = wmma_bf16(a1, b, f10);
    b = load_frag(Wfb, ncol1, k0, lane);
    f01 = wmma_bf16(a0, b, f01); f11 = wmma_bf16(a1, b, f11);
    b = load_frag(Wib, ncol0, k0, lane);
    i00 = wmma_bf16(a0, b, i00); i10 = wmma_bf16(a1, b, i10);
    b = load_frag(Wib, ncol1, k0, lane);
    i01 = wmma_bf16(a0, b, i01); i11 = wmma_bf16(a1, b, i11);
    b = load_frag(Wgb, ncol0, k0, lane);
    g00 = wmma_bf16(a0, b, g00); g10 = wmma_bf16(a1, b, g10);
    b = load_frag(Wgb, ncol1, k0, lane);
    g01 = wmma_bf16(a0, b, g01); g11 = wmma_bf16(a1, b, g11);
  }

  store_fig(f00, i00, g00, mrow0, ncol0, lane, F, INP, G);
  store_fig(f01, i01, g01, mrow0, ncol1, lane, F, INP, G);
  store_fig(f10, i10, g10, mrow1, ncol0, lane, F, INP, G);
  store_fig(f11, i11, g11, mrow1, ncol1, lane, F, INP, G);
}

// ---------------- kernel 2a: per-chunk local scan ----------------
__global__ __launch_bounds__(256) void hgrn_scan_local(
    const float* __restrict__ F, const float* __restrict__ INP,
    float* __restrict__ Fprod, float* __restrict__ Hend) {
  const int tid = blockIdx.x * 256 + threadIdx.x;      // B*NC*H threads
  const int h  = tid & (Hn - 1);
  const int bc = tid >> 10;
  const int c  = bc & (NC - 1);
  const int b  = bc >> 4;
  size_t idx = ((size_t)b * Sn + (size_t)c * CHUNK) * Hn + h;
  float hl = 0.0f, fp = 1.0f;
#pragma unroll 4
  for (int t = 0; t < CHUNK; ++t, idx += Hn) {
    const float f = F[idx];
    hl = f * hl + INP[idx];
    fp *= f;
  }
  Fprod[tid] = fp;     // layout [b, c, h]
  Hend[tid]  = hl;
}

// ---------------- kernel 2b: carry prefix across chunks ----------------
__global__ __launch_bounds__(256) void hgrn_scan_carry(
    const float* __restrict__ Fprod, const float* __restrict__ Hend,
    float* __restrict__ Carry) {
  const int tid = blockIdx.x * 256 + threadIdx.x;      // B*H threads
  const int h = tid & (Hn - 1);
  const int b = tid >> 10;
  float hc = 0.0f;
#pragma unroll
  for (int c = 0; c < NC; ++c) {
    const size_t idx = ((size_t)b * NC + c) * Hn + h;
    Carry[idx] = hc;
    hc = Fprod[idx] * hc + Hend[idx];
  }
}

// ---------------- kernel 2c: apply carries, emit u = g*h (bf16) ----------------
__global__ __launch_bounds__(256) void hgrn_scan_apply(
    const float* __restrict__ F, const float* __restrict__ INP,
    const float* __restrict__ Carry, const __bf16* __restrict__ G,
    __bf16* __restrict__ U) {
  const int tid = blockIdx.x * 256 + threadIdx.x;      // B*NC*H threads
  const int h  = tid & (Hn - 1);
  const int bc = tid >> 10;
  const int c  = bc & (NC - 1);
  const int b  = bc >> 4;
  size_t idx = ((size_t)b * Sn + (size_t)c * CHUNK) * Hn + h;
  float hl = Carry[tid];
#pragma unroll 4
  for (int t = 0; t < CHUNK; ++t, idx += Hn) {
    hl = F[idx] * hl + INP[idx];
    U[idx] = (__bf16)((float)G[idx] * hl);
  }
}

// ---------------- kernel 3: output projection (u @ Wo^T) ----------------
// Wave tile: 32 rows x 64 cols (8 accumulators); 12 b128 loads per 8 wmma.
__global__ __launch_bounds__(128) void hgrn_outproj(
    const __bf16* __restrict__ U, const __bf16* __restrict__ Wob,
    float* __restrict__ out) {
  const int lane = threadIdx.x & 31;
  const int wave = threadIdx.x >> 5;
  const int tile = blockIdx.x * 4 + wave;        // over (m-pair, n-quad)
  const int NQUAD = NT_N / 4;                    // 16
  const int mt = tile / NQUAD;
  const int nq = tile - mt * NQUAD;
  const int mrow0 = mt * 32;
  const int mrow1 = mrow0 + 16;
  const int ncol0 = nq * 64;

  v8f c00 = {}, c01 = {}, c02 = {}, c03 = {};
  v8f c10 = {}, c11 = {}, c12 = {}, c13 = {};
  for (int k0 = 0; k0 < Hn; k0 += 32) {
    v16bf a0 = load_frag(U, mrow0, k0, lane);
    v16bf a1 = load_frag(U, mrow1, k0, lane);
    v16bf b;
    b = load_frag(Wob, ncol0,      k0, lane);
    c00 = wmma_bf16(a0, b, c00); c10 = wmma_bf16(a1, b, c10);
    b = load_frag(Wob, ncol0 + 16, k0, lane);
    c01 = wmma_bf16(a0, b, c01); c11 = wmma_bf16(a1, b, c11);
    b = load_frag(Wob, ncol0 + 32, k0, lane);
    c02 = wmma_bf16(a0, b, c02); c12 = wmma_bf16(a1, b, c12);
    b = load_frag(Wob, ncol0 + 48, k0, lane);
    c03 = wmma_bf16(a0, b, c03); c13 = wmma_bf16(a1, b, c13);
  }

  const int r    = lane & 15;
  const int half = lane >> 4;
#pragma unroll
  for (int j = 0; j < 8; ++j) {
    const size_t rb0 = (size_t)(mrow0 + half * 8 + j) * Hn + r;
    out[rb0 + ncol0]      = c00[j];
    out[rb0 + ncol0 + 16] = c01[j];
    out[rb0 + ncol0 + 32] = c02[j];
    out[rb0 + ncol0 + 48] = c03[j];
    const size_t rb1 = (size_t)(mrow1 + half * 8 + j) * Hn + r;
    out[rb1 + ncol0]      = c10[j];
    out[rb1 + ncol0 + 16] = c11[j];
    out[rb1 + ncol0 + 32] = c12[j];
    out[rb1 + ncol0 + 48] = c13[j];
  }
}

// ---------------- host launcher ----------------
extern "C" void kernel_launch(void* const* d_in, const int* in_sizes, int n_in,
                              void* d_out, int out_size, void* d_ws, size_t ws_size,
                              hipStream_t stream) {
  const float* x  = (const float*)d_in[0];
  const float* Wi = (const float*)d_in[1];
  const float* Wf = (const float*)d_in[2];
  const float* Wg = (const float*)d_in[3];
  const float* Wo = (const float*)d_in[4];
  float* out = (float*)d_out;

  char* ws = (char*)d_ws;
  float*  F     = (float*)(ws + OFF_F);
  float*  INP   = (float*)(ws + OFF_I);
  __bf16* G     = (__bf16*)(ws + OFF_G);
  __bf16* U     = (__bf16*)(ws + OFF_U);
  __bf16* Xb    = (__bf16*)(ws + OFF_XB);
  __bf16* Wfb   = (__bf16*)(ws + OFF_WFB);
  __bf16* Wib   = (__bf16*)(ws + OFF_WIB);
  __bf16* Wgb   = (__bf16*)(ws + OFF_WGB);
  __bf16* Wob   = (__bf16*)(ws + OFF_WOB);
  float*  Fprod = (float*)(ws + OFF_FP);
  float*  Hend  = (float*)(ws + OFF_HE);
  float*  Carry = (float*)(ws + OFF_CA);

  // ---- pre-pass: fp32 -> bf16 (x and the 4 weights) ----
  cvt_bf16<<<(BSn * Hn) / (256 * 8), 256, 0, stream>>>(x, Xb);   // 4096 blocks
  cvt_bf16<<<(Hn * Hn) / (256 * 8), 256, 0, stream>>>(Wf, Wfb);  // 512 blocks
  cvt_bf16<<<(Hn * Hn) / (256 * 8), 256, 0, stream>>>(Wi, Wib);
  cvt_bf16<<<(Hn * Hn) / (256 * 8), 256, 0, stream>>>(Wg, Wgb);
  cvt_bf16<<<(Hn * Hn) / (256 * 8), 256, 0, stream>>>(Wo, Wob);

  // ---- fused triple projection: 256 m-pairs x 32 n-pairs, 4 waves/block ----
  hgrn_proj3<<<((NT_M / 2) * (NT_N / 2)) / 4, 128, 0, stream>>>(Xb, Wfb, Wib,
                                                                Wgb, F, INP, G);

  // ---- chunked scan ----
  const int scan_threads = Bn * NC * Hn;                     // 65536
  hgrn_scan_local<<<scan_threads / 256, 256, 0, stream>>>(F, INP, Fprod, Hend);
  hgrn_scan_carry<<<(Bn * Hn) / 256, 256, 0, stream>>>(Fprod, Hend, Carry);
  hgrn_scan_apply<<<scan_threads / 256, 256, 0, stream>>>(F, INP, Carry, G, U);

  // ---- output projection: 256 m-pairs x 16 n-quads, 4 waves/block ----
  hgrn_outproj<<<((NT_M / 2) * (NT_N / 4)) / 4, 128, 0, stream>>>(U, Wob, out);
}
